// Decoder_35777077575918
// MI455X (gfx1250) — matrix operand order repaired and checked by reading, not compiled, not run on hardware
//
#include <hip/hip_runtime.h>
#include <hip/hip_bf16.h>
#include <stdint.h>

// ---------------------------------------------------------------------------
// LAS decoder for MI455X (gfx1250, wave32, WMMA).
//   * Weights repacked once into k-major bf16; activations bf16; fp32
//     accumulation via V_WMMA_F32_16X16X32_BF16; fp32 LSTM cell state.
//   * Each wave computes TN consecutive 16x16 N-tiles sharing one A fragment
//     per k-step (2 A-loads + 16*TN B-loads per TN WMMAs).
//   * LSTM cell fused into the gates GEMM: weights packed gate-interleaved
//     (p = 4*j+gate) so a TN=4 wave owns 16 hidden units x 4 gates; the tile
//     goes through LDS and c/h are updated in-kernel (no global gate tensor).
//     h is double-buffered by step parity to avoid cross-block WAR hazards.
//   * Logits GEMM + log_softmax fused (N=V=64 fits one block per 16 rows).
//   * Working set (~105 MB) is L2-resident (192 MB); per-step weight traffic
//     (~15 MB bf16) never leaves L2 after warmup.
// Shapes hardcoded to the reference: B=64 T=512 H=512 A=128 V=64 L=200.
// ---------------------------------------------------------------------------

#define NB 64
#define NT 512
#define NH 512
#define NA 128
#define NV 64
#define NL 200

typedef unsigned short u16;
typedef __bf16  bf16x16 __attribute__((ext_vector_type(16)));
typedef float   f32x8   __attribute__((ext_vector_type(8)));

__device__ __forceinline__ u16 f2bf(float f) {
  unsigned u = __float_as_uint(f);
  unsigned r = (u + 0x7FFFu + ((u >> 16) & 1u)) >> 16;   // round-nearest-even
  return (u16)r;
}
__device__ __forceinline__ float bf2f(u16 h) { return __uint_as_float(((unsigned)h) << 16); }
__device__ __forceinline__ float bfl(unsigned u) { return __uint_as_float(u << 16); }
__device__ __forceinline__ float bfh(unsigned u) { return __uint_as_float(u & 0xffff0000u); }
__device__ __forceinline__ float sigm(float x) { return 1.0f / (1.0f + __expf(-x)); }

union FragBF { uint4 u[2]; bf16x16 v; };

struct Seg { const u16* A; const u16* Wt; int K; int lda; };

// Shared WMMA mainloop: accumulate TN 16x16 tiles at (m0, n0..n0+TN*16).
template <int TN>
__device__ __forceinline__ void gemm_core(Seg s0, Seg s1, Seg s2, int nseg, int N,
                                          int m0, int n0, int lane, f32x8 acc[TN]) {
  const int mrow = lane & 15;       // A-frag row within tile
  const int kg8  = (lane >> 4) * 8; // K sub-group offset
  for (int s = 0; s < nseg; ++s) {
    Seg sg = (s == 0) ? s0 : ((s == 1) ? s1 : s2);
    const u16* Arow = sg.A + (size_t)(m0 + mrow) * sg.lda;
    for (int k0 = 0; k0 < sg.K; k0 += 32) {
      // A 16x32 bf16 layout: j=0..7 -> K=k0+kg8+j ; j=8..15 -> K=k0+16+kg8+(j-8)
      FragBF a;
      const u16* pa = Arow + k0 + kg8;
      a.u[0] = *(const uint4*)pa;
      a.u[1] = *(const uint4*)(pa + 16);
      // B 32x16 bf16 layout: lane -> N=lane%16 ; j=0..7 -> K=k0+kg8+j ; j=8..15 -> K=k0+16+kg8+(j-8)
      FragBF b[TN];
      const u16* pb = sg.Wt + (size_t)(k0 + kg8) * N + n0 + mrow;
#pragma unroll
      for (int t = 0; t < TN; ++t) {
        u16 tmp[16];
#pragma unroll
        for (int j = 0; j < 16; ++j)
          tmp[j] = pb[(size_t)((j & 7) + ((j >> 3) * 16)) * N + t * 16];
        __builtin_memcpy(&b[t], tmp, sizeof(tmp));
      }
#pragma unroll
      for (int t = 0; t < TN; ++t)
        acc[t] = __builtin_amdgcn_wmma_f32_16x16x32_bf16(
            false, a.v, false, b[t].v, (short)0, acc[t], false, false);
    }
  }
}

// ---------------- generic GEMM with bias/relu + f32/bf16 stores ------------
template <int TN>
__global__ void wmma_gemm(Seg s0, Seg s1, Seg s2, int nseg, int M, int N,
                          const float* __restrict__ bias, int relu,
                          float* __restrict__ Cf, u16* __restrict__ Cbf) {
  const int lane = threadIdx.x & 31;
  const int wave = threadIdx.x >> 5;
  const int n0 = (blockIdx.x * 4 + wave) * (TN * 16);
  const int m0 = blockIdx.y * 16;
  f32x8 acc[TN];
#pragma unroll
  for (int t = 0; t < TN; ++t) acc[t] = f32x8{};
  gemm_core<TN>(s0, s1, s2, nseg, N, m0, n0, lane, acc);
  const int rbase = m0 + 8 * (lane >> 4);
#pragma unroll
  for (int t = 0; t < TN; ++t) {
    const int col = n0 + t * 16 + (lane & 15);
    const float bb = bias ? bias[col] : 0.0f;
#pragma unroll
    for (int g = 0; g < 8; ++g) {
      float v = acc[t][g] + bb;
      if (relu) v = fmaxf(v, 0.0f);
      size_t idx = (size_t)(rbase + g) * N + col;
      if (Cf)  Cf[idx] = v;
      if (Cbf) Cbf[idx] = f2bf(v);
    }
  }
}

// ---------------- gates GEMM with fused LSTM cell --------------------------
// Packed column p = 4*j + gate (gate: 0=i 1=f 2=g 3=o). A TN=4 wave owns
// packed cols [n0, n0+64) = hidden units [hblk, hblk+16) x 4 gates.
// Tile staged in LDS; each lane then updates 8 cells (c fp32 RMW, h -> bf16).
__global__ void wmma_gemm_lstm(Seg s0, Seg s1, Seg s2, int nseg,
                               const float* __restrict__ biasp,
                               float* __restrict__ cstate, u16* __restrict__ hout) {
  const int lane = threadIdx.x & 31;
  const int wave = threadIdx.x >> 5;
  const int n0 = (blockIdx.x * 4 + wave) * 64;
  const int m0 = blockIdx.y * 16;
  f32x8 acc[4];
#pragma unroll
  for (int t = 0; t < 4; ++t) acc[t] = f32x8{};
  gemm_core<4>(s0, s1, s2, nseg, 4 * NH, m0, n0, lane, acc);

  __shared__ float gs[4][16][64];
  const int lr = 8 * (lane >> 4);
#pragma unroll
  for (int t = 0; t < 4; ++t) {
    const int lc = t * 16 + (lane & 15);
    const float bb = biasp[n0 + lc];
#pragma unroll
    for (int g = 0; g < 8; ++g) gs[wave][lr + g][lc] = acc[t][g] + bb;
  }
  __syncthreads();
  const int hblk = (blockIdx.x * 4 + wave) * 16;
#pragma unroll
  for (int q = 0; q < 8; ++q) {
    const int cid = lane * 8 + q;          // 256 cells per wave
    const int r = cid >> 4, jj = cid & 15;
    const float gi = gs[wave][r][jj * 4 + 0];
    const float gf = gs[wave][r][jj * 4 + 1];
    const float gg = gs[wave][r][jj * 4 + 2];
    const float go = gs[wave][r][jj * 4 + 3];
    const size_t idx = (size_t)(m0 + r) * NH + hblk + jj;
    const float cn = sigm(gf) * cstate[idx] + sigm(gi) * tanhf(gg);
    cstate[idx] = cn;
    hout[idx] = f2bf(sigm(go) * tanhf(cn));
  }
}

// ---------------- logits GEMM with fused log-softmax (N = V = 64) ----------
__global__ void wmma_gemm_logsm(Seg s0, const float* __restrict__ bias,
                                float* __restrict__ out, int l) {
  const int lane = threadIdx.x & 31;
  const int wave = threadIdx.x >> 5;
  const int n0 = wave * 16;           // 4 waves cover all 64 cols
  const int m0 = blockIdx.y * 16;     // grid (1, 4): 16 batch rows per block
  f32x8 acc[1];
  acc[0] = f32x8{};
  Seg z{nullptr, nullptr, 0, 0};
  gemm_core<1>(s0, z, z, 1, NV, m0, n0, lane, acc);

  __shared__ float ls[16][NV];
  const int lr = 8 * (lane >> 4);
  const int lc = n0 + (lane & 15);
  const float bb = bias[lc];
#pragma unroll
  for (int g = 0; g < 8; ++g) ls[lr + g][lc] = acc[0][g] + bb;
  __syncthreads();
  if (threadIdx.x < 16) {
    const int r = (int)threadIdx.x;
    const int b = m0 + r;
    float mx = -1e30f;
    for (int v = 0; v < NV; ++v) mx = fmaxf(mx, ls[r][v]);
    float sum = 0.0f;
    for (int v = 0; v < NV; ++v) sum += __expf(ls[r][v] - mx);
    const float lg = __logf(sum);
    float* o = out + ((size_t)b * NL + l) * NV;
    for (int v = 0; v < NV; ++v) o[v] = ls[r][v] - mx - lg;
  }
}

// ------------------------- small utility kernels ---------------------------
__global__ void k_f32_to_bf16(const float* __restrict__ x, u16* __restrict__ y, size_t n) {
  size_t i = (size_t)blockIdx.x * blockDim.x + threadIdx.x;
  if (i < n) y[i] = f2bf(x[i]);
}

// W[N,K] row-major fp32 -> Wt[K,N] bf16 (k-major)
__global__ void k_pack_wT(const float* __restrict__ W, u16* __restrict__ Wt, int N, int K) {
  size_t i = (size_t)blockIdx.x * blockDim.x + threadIdx.x;
  if (i >= (size_t)N * K) return;
  int k = (int)(i / N), n = (int)(i % N);
  Wt[i] = f2bf(W[(size_t)n * K + k]);
}

// W[2048,K] -> Wt[K,2048] bf16, gate-interleaved: p = 4*j + gate
__global__ void k_pack_ifgo(const float* __restrict__ W, u16* __restrict__ Wt, int K) {
  size_t i = (size_t)blockIdx.x * blockDim.x + threadIdx.x;
  if (i >= (size_t)K * 4 * NH) return;
  int k = (int)(i / (4 * NH)), p = (int)(i % (4 * NH));
  int n = (p & 3) * NH + (p >> 2);
  Wt[i] = f2bf(W[(size_t)n * K + k]);
}

__global__ void k_pack_bias_ifgo(const float* __restrict__ bi, const float* __restrict__ bh,
                                 float* __restrict__ o) {
  int p = blockIdx.x * blockDim.x + threadIdx.x;
  if (p < 4 * NH) {
    int n = (p & 3) * NH + (p >> 2);
    o[p] = bi[n] + bh[n];
  }
}

// emb[l,b,h] = bf16( Wemb[Y[b,l], h] )
__global__ void k_gather_emb(const float* __restrict__ Wemb, const int* __restrict__ Y,
                             u16* __restrict__ emb) {
  size_t i = (size_t)blockIdx.x * blockDim.x + threadIdx.x;
  if (i >= (size_t)NL * NB * NH) return;
  int h = (int)(i % NH);
  int b = (int)((i / NH) % NB);
  int l = (int)(i / ((size_t)NB * NH));
  emb[i] = f2bf(Wemb[(size_t)Y[b * NL + l] * NH + h]);
}

__global__ void k_init_state(const float* __restrict__ h0, const float* __restrict__ c0,
                             u16* __restrict__ hbf, float* __restrict__ c,
                             u16* __restrict__ ctx) {
  int i = blockIdx.x * blockDim.x + threadIdx.x;
  if (i < 3 * NB * NH) {
    int j = i & (NH - 1);
    hbf[i] = f2bf(h0[j]);
    c[i]   = c0[j];
  }
  if (i < NB * 1024) ctx[i] = 0;  // bf16(0) == 0
}

// ---------------- attention: energy + masked softmax -----------------------
__global__ void k_attn_energy(const float* __restrict__ q, const u16* __restrict__ key,
                              const int* __restrict__ flens, u16* __restrict__ att) {
  const int b = blockIdx.x, t = threadIdx.x;
  __shared__ float qs[NA];
  __shared__ float red[NT];
  if (t < NA) qs[t] = q[b * NA + t];
  __syncthreads();
  const uint4* kp = (const uint4*)(key + ((size_t)b * NT + t) * NA);
  float e = 0.0f;
#pragma unroll
  for (int i = 0; i < NA / 8; ++i) {
    uint4 u = kp[i];
    e += qs[i * 8 + 0] * bfl(u.x) + qs[i * 8 + 1] * bfh(u.x)
       + qs[i * 8 + 2] * bfl(u.y) + qs[i * 8 + 3] * bfh(u.y)
       + qs[i * 8 + 4] * bfl(u.z) + qs[i * 8 + 5] * bfh(u.z)
       + qs[i * 8 + 6] * bfl(u.w) + qs[i * 8 + 7] * bfh(u.w);
  }
  red[t] = e; __syncthreads();
  for (int s = NT / 2; s > 0; s >>= 1) { if (t < s) red[t] = fmaxf(red[t], red[t + s]); __syncthreads(); }
  float mx = red[0]; __syncthreads();
  const int valid = flens[b] >> 3;                 // frame_lens // 8
  float w = (t < valid) ? __expf(e - mx) : 0.0f;   // mask + renorm == masked softmax
  red[t] = w; __syncthreads();
  for (int s = NT / 2; s > 0; s >>= 1) { if (t < s) red[t] += red[t + s]; __syncthreads(); }
  att[(size_t)b * NT + t] = f2bf(w / red[0]);
}

// ctx[b,e] = sum_t att[b,t] * Lbf[b,t,e]
__global__ void k_attn_ctx(const u16* __restrict__ att, const u16* __restrict__ Lbf,
                           u16* __restrict__ ctx) {
  const int b = blockIdx.x, tid = threadIdx.x;
  __shared__ float as[NT];
  for (int t = tid; t < NT; t += 256) as[t] = bf2f(att[(size_t)b * NT + t]);
  __syncthreads();
  const int e0 = tid * 4;
  const u16* base = Lbf + (size_t)b * NT * 1024 + e0;
  float a0 = 0, a1 = 0, a2 = 0, a3 = 0;
  for (int t = 0; t < NT; ++t) {
    float a = as[t];
    uint2 u = *(const uint2*)(base + (size_t)t * 1024);
    a0 += a * bfl(u.x); a1 += a * bfh(u.x);
    a2 += a * bfl(u.y); a3 += a * bfh(u.y);
  }
  u16* o = ctx + (size_t)b * 1024 + e0;
  o[0] = f2bf(a0); o[1] = f2bf(a1); o[2] = f2bf(a2); o[3] = f2bf(a3);
}

// ---------------------------------------------------------------------------
static inline void gemm(hipStream_t st, int TN, Seg s0, Seg s1, Seg s2, int nseg,
                        int M, int N, const float* bias, int relu,
                        float* Cf, u16* Cbf) {
  dim3 grid(N / (64 * TN), M / 16);
  if (TN == 2)      wmma_gemm<2><<<grid, 128, 0, st>>>(s0, s1, s2, nseg, M, N, bias, relu, Cf, Cbf);
  else              wmma_gemm<4><<<grid, 128, 0, st>>>(s0, s1, s2, nseg, M, N, bias, relu, Cf, Cbf);
}

extern "C" void kernel_launch(void* const* d_in, const int* in_sizes, int n_in,
                              void* d_out, int out_size, void* d_ws, size_t ws_size,
                              hipStream_t stream) {
  (void)in_sizes; (void)n_in; (void)out_size; (void)ws_size;
  const float* listener = (const float*)d_in[0];
  const int*   Yinput   = (const int*)d_in[1];
  const int*   flens    = (const int*)d_in[2];
  const float* Wemb  = (const float*)d_in[4];
  const float* bproj = (const float*)d_in[5];
  const float* W_ih1 = (const float*)d_in[6];  const float* W_hh1 = (const float*)d_in[7];
  const float* b_ih1 = (const float*)d_in[8];  const float* b_hh1 = (const float*)d_in[9];
  const float* W_ih2 = (const float*)d_in[10]; const float* W_hh2 = (const float*)d_in[11];
  const float* b_ih2 = (const float*)d_in[12]; const float* b_hh2 = (const float*)d_in[13];
  const float* W_ih3 = (const float*)d_in[14]; const float* W_hh3 = (const float*)d_in[15];
  const float* b_ih3 = (const float*)d_in[16]; const float* b_hh3 = (const float*)d_in[17];
  const float* Wq = (const float*)d_in[18];  const float* bq = (const float*)d_in[19];
  const float* Wk = (const float*)d_in[20];  const float* bk = (const float*)d_in[21];
  const float* Wmlp = (const float*)d_in[22]; const float* bmlp = (const float*)d_in[23];
  const float* h0 = (const float*)d_in[24];  const float* c0 = (const float*)d_in[25];
  float* out = (float*)d_out;

  // ---- workspace bump allocator ----
  char* ws = (char*)d_ws; size_t off = 0;
  auto alloc = [&](size_t bytes) -> void* {
    void* p = ws + off; off = (off + bytes + 255) & ~(size_t)255; return p;
  };
  u16* Lbf    = (u16*)alloc((size_t)NB * NT * 1024 * 2);
  u16* keybf  = (u16*)alloc((size_t)NB * NT * NA * 2);
  u16* embbf  = (u16*)alloc((size_t)NL * NB * NH * 2);
  u16* Wih1T  = (u16*)alloc((size_t)1536 * 2048 * 2);   // gate-interleaved
  u16* Whh1T  = (u16*)alloc((size_t)512 * 2048 * 2);    // gate-interleaved
  u16* Wih2T  = (u16*)alloc((size_t)512 * 2048 * 2);
  u16* Whh2T  = (u16*)alloc((size_t)512 * 2048 * 2);
  u16* Wih3T  = (u16*)alloc((size_t)512 * 2048 * 2);
  u16* Whh3T  = (u16*)alloc((size_t)512 * 2048 * 2);
  u16* WqT    = (u16*)alloc((size_t)512 * NA * 2);
  u16* WkT    = (u16*)alloc((size_t)1024 * NA * 2);
  u16* WmlpT  = (u16*)alloc((size_t)1536 * 512 * 2);
  u16* WembT  = (u16*)alloc((size_t)512 * NV * 2);
  float* bias1 = (float*)alloc(2048 * 4);  // gate-interleaved packed biases
  float* bias2 = (float*)alloc(2048 * 4);
  float* bias3 = (float*)alloc(2048 * 4);
  u16* hA      = (u16*)alloc((size_t)3 * NB * NH * 2);  // h double buffers
  u16* hB      = (u16*)alloc((size_t)3 * NB * NH * 2);
  float* cst   = (float*)alloc((size_t)3 * NB * NH * 4);
  float* qbuf  = (float*)alloc((size_t)NB * NA * 4);
  u16* attbuf  = (u16*)alloc((size_t)NB * NT * 2);
  u16* ctxbf   = (u16*)alloc((size_t)NB * 1024 * 2);
  u16* mlpbf   = (u16*)alloc((size_t)NB * NH * 2);

  float* c1 = cst; float* c2 = cst + NB * NH; float* c3 = cst + 2 * NB * NH;

  auto cdiv = [](size_t a, size_t b) { return (unsigned)((a + b - 1) / b); };

  // ---- one-time precompute (every call: deterministic) ----
  { size_t n = (size_t)NB * NT * 1024;
    k_f32_to_bf16<<<cdiv(n, 256), 256, 0, stream>>>(listener, Lbf, n); }
  k_pack_ifgo<<<cdiv((size_t)1536 * 2048, 256), 256, 0, stream>>>(W_ih1, Wih1T, 1536);
  k_pack_ifgo<<<cdiv((size_t)512 * 2048, 256), 256, 0, stream>>>(W_hh1, Whh1T, 512);
  k_pack_ifgo<<<cdiv((size_t)512 * 2048, 256), 256, 0, stream>>>(W_ih2, Wih2T, 512);
  k_pack_ifgo<<<cdiv((size_t)512 * 2048, 256), 256, 0, stream>>>(W_hh2, Whh2T, 512);
  k_pack_ifgo<<<cdiv((size_t)512 * 2048, 256), 256, 0, stream>>>(W_ih3, Wih3T, 512);
  k_pack_ifgo<<<cdiv((size_t)512 * 2048, 256), 256, 0, stream>>>(W_hh3, Whh3T, 512);
  k_pack_wT<<<cdiv((size_t)NA * 512, 256), 256, 0, stream>>>(Wq, WqT, NA, 512);
  k_pack_wT<<<cdiv((size_t)NA * 1024, 256), 256, 0, stream>>>(Wk, WkT, NA, 1024);
  k_pack_wT<<<cdiv((size_t)512 * 1536, 256), 256, 0, stream>>>(Wmlp, WmlpT, 512, 1536);
  k_pack_wT<<<cdiv((size_t)NV * 512, 256), 256, 0, stream>>>(Wemb, WembT, NV, 512);
  k_pack_bias_ifgo<<<8, 256, 0, stream>>>(b_ih1, b_hh1, bias1);
  k_pack_bias_ifgo<<<8, 256, 0, stream>>>(b_ih2, b_hh2, bias2);
  k_pack_bias_ifgo<<<8, 256, 0, stream>>>(b_ih3, b_hh3, bias3);
  { size_t n = (size_t)NL * NB * NH;
    k_gather_emb<<<cdiv(n, 256), 256, 0, stream>>>(Wemb, Yinput, embbf); }
  k_init_state<<<cdiv(3 * NB * NH, 256), 256, 0, stream>>>(h0, c0, hA, cst, ctxbf);

  Seg z{nullptr, nullptr, 0, 0};
  // key_feat = relu(listener @ Wk^T + bk): [B*T,1024]@[1024,128] -> bf16
  gemm(stream, 2, Seg{Lbf, WkT, 1024, 1024}, z, z, 1, NB * NT, NA, bk, 1, nullptr, keybf);

  auto attend = [&](const u16* h3) {
    gemm(stream, 2, Seg{h3, WqT, 512, 512}, z, z, 1, NB, NA, bq, 1, qbuf, nullptr);
    k_attn_energy<<<NB, NT, 0, stream>>>(qbuf, keybf, flens, attbuf);
    k_attn_ctx<<<NB, 256, 0, stream>>>(attbuf, Lbf, ctxbf);
  };
  attend(hA + 2 * NB * NH);  // context0 from h0

  const dim3 lgrid(8, 4);    // N=2048 / (64*4), M=64/16
  for (int l = 0; l < NL; ++l) {
    u16* hr = (l & 1) ? hB : hA;   // recurrent inputs (previous step)
    u16* hw = (l & 1) ? hA : hB;   // outputs of this step
    u16* h1r = hr;               u16* h1w = hw;
    u16* h2r = hr + NB * NH;     u16* h2w = hw + NB * NH;
    u16* h3r = hr + 2 * NB * NH; u16* h3w = hw + 2 * NB * NH;
    const u16* emb_l = embbf + (size_t)l * NB * NH;

    // layer 1: gates = [emb|ctx]@W_ih1^T + h1@W_hh1^T + b ; fused cell
    wmma_gemm_lstm<<<lgrid, 128, 0, stream>>>(
        Seg{emb_l, Wih1T, 512, 512},
        Seg{ctxbf, Wih1T + (size_t)512 * 2048, 1024, 1024},
        Seg{h1r, Whh1T, 512, 512}, 3, bias1, c1, h1w);
    // layer 2
    wmma_gemm_lstm<<<lgrid, 128, 0, stream>>>(
        Seg{h1w, Wih2T, 512, 512}, Seg{h2r, Whh2T, 512, 512}, z,
        2, bias2, c2, h2w);
    // layer 3
    wmma_gemm_lstm<<<lgrid, 128, 0, stream>>>(
        Seg{h2w, Wih3T, 512, 512}, Seg{h3r, Whh3T, 512, 512}, z,
        2, bias3, c3, h3w);
    // attention with new h3
    attend(h3w);
    // mlp = relu([h3|ctx] @ Wmlp^T + bmlp) -> bf16
    gemm(stream, 4, Seg{h3w, WmlpT, 512, 512},
                    Seg{ctxbf, WmlpT + (size_t)512 * 512, 1024, 1024}, z,
         2, NB, 512, bmlp, 1, nullptr, mlpbf);
    // logits = mlp @ Wemb^T + b_proj ; fused log_softmax into d_out
    wmma_gemm_logsm<<<dim3(1, 4), 128, 0, stream>>>(
        Seg{mlpbf, WembT, 512, 512}, bproj, out, l);
  }
}
